// Bi_mamba_block_67327907332574
// MI455X (gfx1250) — matrix operand order repaired and checked by reading, not compiled, not run on hardware
//
#include <hip/hip_runtime.h>
#include <hip/hip_bf16.h>
#include <math.h>

// ---------------------------------------------------------------------------
// Types for CDNA5 WMMA / TDM (gfx1250, wave32)
// ---------------------------------------------------------------------------
typedef __bf16 bf16;
typedef __attribute__((ext_vector_type(16))) __bf16 v16bf;
typedef __attribute__((ext_vector_type(8)))  __bf16 v8bf;
typedef __attribute__((ext_vector_type(8)))  float  v8f;
typedef __attribute__((ext_vector_type(4)))  unsigned int u32x4;
typedef __attribute__((ext_vector_type(8)))  int i32x8;
typedef __attribute__((ext_vector_type(4)))  int i32x4;

// Problem constants (match reference)
#define DMODEL 768
#define DINNER 1536
#define DCONV  4
#define DSTATE 16
#define DTRANK 48
#define DTPAD  64       // dt_rank padded to multiple of 32 for WMMA K
#define XPN    128      // x_proj output (80) padded to multiple of 128
#define BATCH  2
#define SEQ    2048
#define MTOK   (BATCH*SEQ)   // 4096 rows

// GEMM tiling
#define KCHUNK 64       // K staged per TDM transfer
#define BM     128      // block M tile (4 waves x 32)
#define BN     128      // block N tile (= TDM tile rows, 2 waves x 64)
#define WM     32       // wave M tile
#define WN     64       // wave N tile

// Chunked scan
#define NCHUNK 32
#define CLEN   (SEQ / NCHUNK)   // 64

#if defined(__has_builtin)
#if __has_builtin(__builtin_amdgcn_tensor_load_to_lds) && __has_builtin(__builtin_amdgcn_s_wait_tensorcnt)
#define USE_TDM 1
#endif
#endif
#ifndef USE_TDM
#define USE_TDM 0
#endif

#if USE_TDM
// ---------------------------------------------------------------------------
// Issue one TDM transfer: 2-D tile of BN rows x KCHUNK bf16 cols from a
// row-major [tensor_n, tensor_k] bf16 matrix into LDS (row-major BNxKCHUNK,
// 16 KB). D# layout per CDNA5 ISA ch.8 (group0: count/lds_addr/global_addr/
// type, group1: data_size, tensor dims, tile dims, dim0 stride).
// ---------------------------------------------------------------------------
__device__ __forceinline__
void tdm_load_tile(unsigned lds_off, const bf16* gaddr, int tensor_k, int tensor_n)
{
    unsigned long long ga = (unsigned long long)(uintptr_t)gaddr;
    u32x4 g0;
    g0[0] = 1u;                                            // count=1, user D#
    g0[1] = lds_off;                                       // lds_addr
    g0[2] = (unsigned)(ga & 0xffffffffu);                  // global_addr[31:0]
    g0[3] = (unsigned)((ga >> 32) & 0x01ffffffu) | (2u << 30); // addr[56:32]|type=2
    i32x8 g1;
    g1[0] = (int)(1u << 16);                               // data_size=1 -> 2 bytes
    g1[1] = (int)(((unsigned)tensor_k & 0xffffu) << 16);   // tensor_dim0[15:0]
    g1[2] = (int)((((unsigned)tensor_k >> 16) & 0xffffu) |
                  (((unsigned)tensor_n & 0xffffu) << 16)); // dim0[31:16]|dim1[15:0]
    g1[3] = (int)((((unsigned)tensor_n >> 16) & 0xffffu) |
                  ((unsigned)KCHUNK << 16));               // dim1[31:16]|tile_dim0
    g1[4] = (int)(unsigned)BN;                             // tile_dim1 | tile_dim2=0
    g1[5] = (int)(unsigned)tensor_k;                       // dim0_stride[31:0]
    g1[6] = 0;                                             // stride[47:32]|d1str lo
    g1[7] = 0;
    i32x4 gz = {0, 0, 0, 0};
#if __clang_major__ >= 23
    i32x8 gz8 = {0, 0, 0, 0, 0, 0, 0, 0};
    __builtin_amdgcn_tensor_load_to_lds(g0, g1, gz, gz, gz8, 0);
#else
    __builtin_amdgcn_tensor_load_to_lds(g0, g1, gz, gz, 0);
#endif
}
#endif

// ---------------------------------------------------------------------------
// Generic fp32 -> bf16 convert with row/col zero padding.
// out[r*Cp + c] = (r < Rin && c < Ccopy) ? in[r*Cs + c] : 0
// ---------------------------------------------------------------------------
__global__ __launch_bounds__(256)
void convert_pad_kernel(const float* __restrict__ in, bf16* __restrict__ out,
                        int Rout, int Rin, int Cs, int Ccopy, int Cp)
{
    size_t total = (size_t)Rout * (size_t)Cp;
    for (size_t i = (size_t)blockIdx.x * 256u + threadIdx.x; i < total;
         i += (size_t)gridDim.x * 256u) {
        int c = (int)(i % Cp);
        int r = (int)(i / Cp);
        float v = (r < Rin && c < Ccopy) ? in[(size_t)r * Cs + c] : 0.0f;
        out[i] = (bf16)v;
    }
}

// ---------------------------------------------------------------------------
// LayerNorm: one block (256 thr) per row of length N (=768), output bf16
// ---------------------------------------------------------------------------
__global__ __launch_bounds__(256)
void layernorm_kernel(const float* __restrict__ in, const float* __restrict__ g,
                      const float* __restrict__ b, bf16* __restrict__ out,
                      int N)
{
    __shared__ float red[256];
    int row = blockIdx.x;
    const float* x = in + (size_t)row * N;

    float s = 0.f;
    for (int c = threadIdx.x; c < N; c += 256) s += x[c];
    red[threadIdx.x] = s;
    __syncthreads();
    for (int off = 128; off > 0; off >>= 1) {
        if (threadIdx.x < off) red[threadIdx.x] += red[threadIdx.x + off];
        __syncthreads();
    }
    float mu = red[0] / (float)N;
    __syncthreads();

    float v = 0.f;
    for (int c = threadIdx.x; c < N; c += 256) { float d = x[c] - mu; v += d * d; }
    red[threadIdx.x] = v;
    __syncthreads();
    for (int off = 128; off > 0; off >>= 1) {
        if (threadIdx.x < off) red[threadIdx.x] += red[threadIdx.x + off];
        __syncthreads();
    }
    float rstd = rsqrtf(red[0] / (float)N + 1e-5f);

    for (int c = threadIdx.x; c < N; c += 256)
        out[(size_t)row * N + c] = (bf16)((x[c] - mu) * rstd * g[c] + b[c]);
}

// ---------------------------------------------------------------------------
// BF16 WMMA GEMM with TDM-staged B tiles:
//   C[M,N] = A[M,K] @ W[N,K]^T   (row-major A, row-major W)
// Block: 8 waves as 4(M) x 2(N) -> 128 x 128. Wave: 32(M) x 64(N),
// 8 f32 accumulators (each LDS B fragment feeds two WMMAs).
// B tile (128 N-rows x 64 K) staged to LDS by wave 0 via tensor_load_to_lds,
// double buffered; waves read fragments from LDS (ds_load_b128).
// A fragments stream straight from global (per-wave unique rows).
// Epilogue: optional bias, act (0:none 1:silu 2:gelu 3:softplus),
// optional fp32 add_src, optional accumulate-into out_f32, optional bf16 out.
// ---------------------------------------------------------------------------
__global__ __launch_bounds__(256)
void gemm_bf16_wmma(const bf16* __restrict__ A, const bf16* __restrict__ W,
                    int M, int N, int K,
                    float* __restrict__ out_f32, bf16* __restrict__ out_bf16,
                    const float* __restrict__ bias, int act,
                    const float* __restrict__ add_src, int accumulate)
{
    __shared__ __align__(16) bf16 sB[2][BN * KCHUNK];   // 2 x 16 KB

    const int wave = threadIdx.x >> 5;     // 0..7
    const int lane = threadIdx.x & 31;
    const int r    = lane & 15;
    const int half = lane >> 4;
    const int wm   = wave >> 1;            // 0..3
    const int wn   = wave & 1;             // 0..1

    const int bn0 = blockIdx.y * BN;
    const int m0  = blockIdx.x * BM + wm * WM;
    const int n0  = bn0 + wn * WN;

    v8f acc[2][4];
#pragma unroll
    for (int gph = 0; gph < 2; ++gph)
#pragma unroll
        for (int j = 0; j < 4; ++j)
#pragma unroll
            for (int i = 0; i < 8; ++i) acc[gph][j][i] = 0.0f;

    // A: lane(0..15) -> row +r, K {0..7,16..23}; lane(16..31) -> K {8..15,24..31}
    const bf16* pA0 = A + (size_t)(m0 + r) * K + half * 8;
    const bf16* pA1 = A + (size_t)(m0 + 16 + r) * K + half * 8;
    const int nstages = K / KCHUNK;

#if USE_TDM
    const unsigned lds0 = (unsigned)(uintptr_t)(&sB[0][0]);
    if (wave == 0)
        tdm_load_tile(lds0, W + (size_t)bn0 * K, K, N);
#endif

    for (int s = 0; s < nstages; ++s) {
        const int cur = s & 1;
#if USE_TDM
        if (wave == 0) __builtin_amdgcn_s_wait_tensorcnt(0);
        __syncthreads();
        if (wave == 0 && s + 1 < nstages)
            tdm_load_tile(lds0 + (unsigned)(1 - cur) * (BN * KCHUNK * 2),
                          W + (size_t)bn0 * K + (size_t)(s + 1) * KCHUNK, K, N);
#else
        // Fallback: cooperative copy, 256 threads x 64 B each
        __syncthreads();
        {
            int row = threadIdx.x >> 1;                 // 0..127
            int c32 = (threadIdx.x & 1) * 32;
            const v8bf* sp = (const v8bf*)(W + (size_t)(bn0 + row) * K +
                                           (size_t)s * KCHUNK + c32);
            v8bf* dp = (v8bf*)(&sB[cur][row * KCHUNK + c32]);
            dp[0] = sp[0]; dp[1] = sp[1]; dp[2] = sp[2]; dp[3] = sp[3];
        }
        __syncthreads();
#endif
        const bf16* sb = &sB[cur][wn * WN * KCHUNK];    // this wave's 64-row half
        const int kb = s * KCHUNK;
#pragma unroll
        for (int kk = 0; kk < KCHUNK; kk += 32) {
            v8bf a0lo = *(const v8bf*)(pA0 + kb + kk);
            v8bf a0hi = *(const v8bf*)(pA0 + kb + kk + 16);
            v8bf a1lo = *(const v8bf*)(pA1 + kb + kk);
            v8bf a1hi = *(const v8bf*)(pA1 + kb + kk + 16);
            v16bf af0, af1;
#pragma unroll
            for (int i = 0; i < 8; ++i) {
                af0[i] = a0lo[i]; af0[i + 8] = a0hi[i];
                af1[i] = a1lo[i]; af1[i + 8] = a1hi[i];
            }
#pragma unroll
            for (int j = 0; j < 4; ++j) {
                // B: lane(0..15) -> col +j*16+r, K {kk..kk+7, kk+16..kk+23};
                //    lane(16..31) -> K {kk+8..kk+15, kk+24..kk+31}
                const bf16* pb = sb + (j * 16 + r) * KCHUNK + kk + half * 8;
                v8bf blo = *(const v8bf*)(pb);
                v8bf bhi = *(const v8bf*)(pb + 16);
                v16bf bfrag;
#pragma unroll
                for (int i = 0; i < 8; ++i) { bfrag[i] = blo[i]; bfrag[i + 8] = bhi[i]; }
                acc[0][j] = __builtin_amdgcn_wmma_f32_16x16x32_bf16(
                    false, af0, false, bfrag, (short)0, acc[0][j], false, false);
                acc[1][j] = __builtin_amdgcn_wmma_f32_16x16x32_bf16(
                    false, af1, false, bfrag, (short)0, acc[1][j], false, false);
            }
        }
        __syncthreads();
    }

    // Epilogue. C/D layout: VGPR i, lanes 0-15 -> (mbase+i, n0+lane),
    // lanes 16-31 -> (mbase+i+8, n0+lane-16).
    const int colb = n0 + r;
#pragma unroll
    for (int gph = 0; gph < 2; ++gph) {
        const int mbase = m0 + gph * 16;
#pragma unroll
        for (int j = 0; j < 4; ++j) {
            const int col = colb + j * 16;
#pragma unroll
            for (int i = 0; i < 8; ++i) {
                const int row = mbase + i + half * 8;
                float v = acc[gph][j][i];
                if (bias) v += bias[col];
                if (act == 1) {                       // silu
                    v = v / (1.0f + __expf(-v));
                } else if (act == 2) {                // exact gelu
                    v = 0.5f * v * (1.0f + erff(v * 0.70710678118f));
                } else if (act == 3) {                // softplus
                    v = (v > 20.0f) ? v : log1pf(__expf(v));
                }
                const size_t o = (size_t)row * N + col;
                if (add_src) v += add_src[o];
                if (out_f32) {
                    if (accumulate) out_f32[o] += v; else out_f32[o] = v;
                }
                if (out_bf16) out_bf16[o] = (bf16)v;
            }
        }
    }
}

// ---------------------------------------------------------------------------
// Causal (dir=0) / anti-causal (dir=1) depthwise conv over time + bias + SiLU.
// Input xz [M,3072] (xh = cols 0..1535). Outputs fp32 and bf16 [M,1536].
// ---------------------------------------------------------------------------
__global__ __launch_bounds__(256)
void conv_silu_kernel(const float* __restrict__ xz, const float* __restrict__ w,
                      const float* __restrict__ bias, float* __restrict__ out_f,
                      bf16* __restrict__ out_b, int dir)
{
    const size_t total = (size_t)MTOK * DINNER;
    for (size_t i = (size_t)blockIdx.x * 256u + threadIdx.x; i < total;
         i += (size_t)gridDim.x * 256u) {
        int d  = (int)(i % DINNER);
        size_t bt = i / DINNER;
        int t  = (int)(bt % SEQ);
        int bb = (int)(bt / SEQ);
        const float* base = xz + ((size_t)bb * SEQ) * (2 * DINNER) + d;
        float acc = bias[d];
#pragma unroll
        for (int k = 0; k < DCONV; ++k) {
            int tt = (dir == 0) ? (t - (DCONV - 1) + k) : (t + (DCONV - 1) - k);
            if (tt >= 0 && tt < SEQ)
                acc += w[d * DCONV + k] * base[(size_t)tt * (2 * DINNER)];
        }
        float s = acc / (1.0f + __expf(-acc));
        out_f[i] = s;
        out_b[i] = (bf16)s;
    }
}

// ---------------------------------------------------------------------------
// Chunked selective scan, 3 phases. Chunk transition operator uses
//   prod_t exp(dt_t * A_n) = exp(A_n * sum_t dt_t).
// xdbl rows are 128 wide: dt 0..47, B 48..63, C 64..79.
// s is the scan-order index; t = s (fwd) or SEQ-1-s (bwd).
// ---------------------------------------------------------------------------
__global__ __launch_bounds__(256)
void scan_phase1(const float* __restrict__ u, const float* __restrict__ dt,
                 const float* __restrict__ A_log, const float* __restrict__ xdbl,
                 float* __restrict__ hend, float* __restrict__ ssum, int dir)
{
    int tid = blockIdx.x * 256 + threadIdx.x;
    if (tid >= BATCH * DINNER * NCHUNK) return;
    int d  = tid % DINNER;
    int ch = (tid / DINNER) % NCHUNK;
    int bb = tid / (DINNER * NCHUNK);

    float An[DSTATE];
#pragma unroll
    for (int n = 0; n < DSTATE; ++n) An[n] = -__expf(A_log[d * DSTATE + n]);

    float h[DSTATE];
#pragma unroll
    for (int n = 0; n < DSTATE; ++n) h[n] = 0.0f;
    float S = 0.0f;

    for (int s = ch * CLEN; s < (ch + 1) * CLEN; ++s) {
        int t = (dir == 0) ? s : (SEQ - 1 - s);
        size_t row = (size_t)bb * SEQ + t;
        float dtv = dt[row * DINNER + d];
        float uv  = u[row * DINNER + d];
        const float* xr = xdbl + row * XPN;
        float du = dtv * uv;
        S += dtv;
#pragma unroll
        for (int n = 0; n < DSTATE; ++n)
            h[n] = h[n] * __expf(dtv * An[n]) + du * xr[DTRANK + n];
    }
    size_t o = ((size_t)(bb * DINNER + d) * NCHUNK + ch) * DSTATE;
#pragma unroll
    for (int n = 0; n < DSTATE; ++n) hend[o + n] = h[n];
    ssum[(size_t)(bb * DINNER + d) * NCHUNK + ch] = S;
}

__global__ __launch_bounds__(256)
void scan_phase2(const float* __restrict__ A_log, const float* __restrict__ hend,
                 const float* __restrict__ ssum, float* __restrict__ h0buf)
{
    int tid = blockIdx.x * 256 + threadIdx.x;
    if (tid >= BATCH * DINNER) return;
    int d  = tid % DINNER;
    int bb = tid / DINNER;

    float An[DSTATE];
#pragma unroll
    for (int n = 0; n < DSTATE; ++n) An[n] = -__expf(A_log[d * DSTATE + n]);

    float h0[DSTATE];
#pragma unroll
    for (int n = 0; n < DSTATE; ++n) h0[n] = 0.0f;

    size_t basebd = (size_t)(bb * DINNER + d) * NCHUNK;
    for (int c = 0; c < NCHUNK; ++c) {
        size_t o = (basebd + c) * DSTATE;
#pragma unroll
        for (int n = 0; n < DSTATE; ++n) h0buf[o + n] = h0[n];
        float S = ssum[basebd + c];
#pragma unroll
        for (int n = 0; n < DSTATE; ++n)
            h0[n] = hend[o + n] + __expf(An[n] * S) * h0[n];
    }
}

__global__ __launch_bounds__(256)
void scan_phase3(const float* __restrict__ u, const float* __restrict__ dt,
                 const float* __restrict__ A_log, const float* __restrict__ Dp,
                 const float* __restrict__ xdbl, const float* __restrict__ h0buf,
                 float* __restrict__ y, int dir)
{
    int tid = blockIdx.x * 256 + threadIdx.x;
    if (tid >= BATCH * DINNER * NCHUNK) return;
    int d  = tid % DINNER;
    int ch = (tid / DINNER) % NCHUNK;
    int bb = tid / (DINNER * NCHUNK);

    float An[DSTATE];
#pragma unroll
    for (int n = 0; n < DSTATE; ++n) An[n] = -__expf(A_log[d * DSTATE + n]);
    const float Dv = Dp[d];

    float h[DSTATE];
    size_t o = ((size_t)(bb * DINNER + d) * NCHUNK + ch) * DSTATE;
#pragma unroll
    for (int n = 0; n < DSTATE; ++n) h[n] = h0buf[o + n];

    for (int s = ch * CLEN; s < (ch + 1) * CLEN; ++s) {
        int t = (dir == 0) ? s : (SEQ - 1 - s);
        size_t row = (size_t)bb * SEQ + t;
        float dtv = dt[row * DINNER + d];
        float uv  = u[row * DINNER + d];
        const float* xr = xdbl + row * XPN;
        float du  = dtv * uv;
        float acc = 0.0f;
#pragma unroll
        for (int n = 0; n < DSTATE; ++n) {
            h[n] = h[n] * __expf(dtv * An[n]) + du * xr[DTRANK + n];
            acc += h[n] * xr[DTRANK + DSTATE + n];
        }
        y[row * DINNER + d] = acc + uv * Dv;
    }
}

// ---------------------------------------------------------------------------
// out = bf16( y * silu(z) ), z = xz[:, 1536 + d]
// ---------------------------------------------------------------------------
__global__ __launch_bounds__(256)
void gate_mul_kernel(const float* __restrict__ y, const float* __restrict__ xz,
                     bf16* __restrict__ out)
{
    const size_t total = (size_t)MTOK * DINNER;
    for (size_t i = (size_t)blockIdx.x * 256u + threadIdx.x; i < total;
         i += (size_t)gridDim.x * 256u) {
        int d = (int)(i % DINNER);
        size_t m = i / DINNER;
        float z = xz[m * (2 * DINNER) + DINNER + d];
        float s = z / (1.0f + __expf(-z));
        out[i] = (bf16)(y[i] * s);
    }
}

// ---------------------------------------------------------------------------
// Host-side launch sequence
// ---------------------------------------------------------------------------
extern "C" void kernel_launch(void* const* d_in, const int* in_sizes, int n_in,
                              void* d_out, int out_size, void* d_ws, size_t ws_size,
                              hipStream_t stream)
{
    (void)in_sizes; (void)n_in; (void)out_size; (void)ws_size;

    const float* x      = (const float*)d_in[0];
    const float* in_w[2]    = {(const float*)d_in[1],  (const float*)d_in[10]};
    const float* conv_w[2]  = {(const float*)d_in[2],  (const float*)d_in[11]};
    const float* conv_b[2]  = {(const float*)d_in[3],  (const float*)d_in[12]};
    const float* xproj_w[2] = {(const float*)d_in[4],  (const float*)d_in[13]};
    const float* dt_w[2]    = {(const float*)d_in[5],  (const float*)d_in[14]};
    const float* dt_b[2]    = {(const float*)d_in[6],  (const float*)d_in[15]};
    const float* A_log[2]   = {(const float*)d_in[7],  (const float*)d_in[16]};
    const float* Dp[2]      = {(const float*)d_in[8],  (const float*)d_in[17]};
    const float* out_w[2]   = {(const float*)d_in[9],  (const float*)d_in[18]};
    const float* ln1_g = (const float*)d_in[19];
    const float* ln1_b = (const float*)d_in[20];
    const float* ln2_g = (const float*)d_in[21];
    const float* ln2_b = (const float*)d_in[22];
    const float* ff_w1 = (const float*)d_in[23];
    const float* ff_b1 = (const float*)d_in[24];
    const float* ff_w2 = (const float*)d_in[25];
    const float* ff_b2 = (const float*)d_in[26];
    float* outp = (float*)d_out;

    // ---- workspace carve-up (256-byte aligned) ----
    char* base = (char*)d_ws;
    size_t off = 0;
    auto alloc = [&](size_t bytes) -> void* {
        off = (off + 255) & ~(size_t)255;
        void* p = (void*)(base + off);
        off += bytes;
        return p;
    };
    bf16* inw_bf[2]   = {(bf16*)alloc((size_t)2*DINNER*DMODEL*2),
                         (bf16*)alloc((size_t)2*DINNER*DMODEL*2)};
    bf16* xprojw_bf[2]= {(bf16*)alloc((size_t)XPN*DINNER*2),
                         (bf16*)alloc((size_t)XPN*DINNER*2)};
    bf16* dtw_bf[2]   = {(bf16*)alloc((size_t)DINNER*DTPAD*2),
                         (bf16*)alloc((size_t)DINNER*DTPAD*2)};
    bf16* outw_bf[2]  = {(bf16*)alloc((size_t)DMODEL*DINNER*2),
                         (bf16*)alloc((size_t)DMODEL*DINNER*2)};
    bf16* ffw1_bf = (bf16*)alloc((size_t)4*DMODEL*DMODEL*2);
    bf16* ffw2_bf = (bf16*)alloc((size_t)DMODEL*4*DMODEL*2);

    bf16*  xn_bf    = (bf16*)alloc((size_t)MTOK*DMODEL*2);
    float* xz_f     = (float*)alloc((size_t)MTOK*2*DINNER*4);
    float* xh_f     = (float*)alloc((size_t)MTOK*DINNER*4);
    bf16*  xh_bf    = (bf16*)alloc((size_t)MTOK*DINNER*2);
    float* xdbl_f   = (float*)alloc((size_t)MTOK*XPN*4);
    bf16*  dtin_bf  = (bf16*)alloc((size_t)MTOK*DTPAD*2);
    float* dt_f     = (float*)alloc((size_t)MTOK*DINNER*4);
    float* y_f      = (float*)alloc((size_t)MTOK*DINNER*4);
    bf16*  ymul_bf  = (bf16*)alloc((size_t)MTOK*DINNER*2);
    float* fo_f     = (float*)alloc((size_t)MTOK*DMODEL*4);
    bf16*  m_bf     = (bf16*)alloc((size_t)MTOK*DMODEL*2);
    bf16*  h1_bf    = (bf16*)alloc((size_t)MTOK*4*DMODEL*2);
    // chunked-scan scratch
    float* hend_f   = (float*)alloc((size_t)BATCH*DINNER*NCHUNK*DSTATE*4);
    float* h0buf_f  = (float*)alloc((size_t)BATCH*DINNER*NCHUNK*DSTATE*4);
    float* ssum_f   = (float*)alloc((size_t)BATCH*DINNER*NCHUNK*4);

    const int EB = 2048;   // grid-stride elementwise blocks
    const int SCAN_B = (BATCH * DINNER * NCHUNK + 255) / 256;   // 384

    // ---- 1. weight conversions to bf16 (with padding where needed) ----
    for (int d = 0; d < 2; ++d) {
        convert_pad_kernel<<<EB, 256, 0, stream>>>(in_w[d],   inw_bf[d],
            2*DINNER, 2*DINNER, DMODEL, DMODEL, DMODEL);
        convert_pad_kernel<<<EB, 256, 0, stream>>>(xproj_w[d], xprojw_bf[d],
            XPN, DTRANK + 2*DSTATE, DINNER, DINNER, DINNER);
        convert_pad_kernel<<<EB, 256, 0, stream>>>(dt_w[d],   dtw_bf[d],
            DINNER, DINNER, DTRANK, DTRANK, DTPAD);
        convert_pad_kernel<<<EB, 256, 0, stream>>>(out_w[d],  outw_bf[d],
            DMODEL, DMODEL, DINNER, DINNER, DINNER);
    }
    convert_pad_kernel<<<EB, 256, 0, stream>>>(ff_w1, ffw1_bf,
        4*DMODEL, 4*DMODEL, DMODEL, DMODEL, DMODEL);
    convert_pad_kernel<<<EB, 256, 0, stream>>>(ff_w2, ffw2_bf,
        DMODEL, DMODEL, 4*DMODEL, 4*DMODEL, 4*DMODEL);

    // ---- 2. LN1 ----
    layernorm_kernel<<<MTOK, 256, 0, stream>>>(x, ln1_g, ln1_b, xn_bf, DMODEL);

    // ---- 3. per-direction Mamba branch (dir 0 = forward, 1 = backward) ----
    for (int dir = 0; dir < 2; ++dir) {
        // in_proj: [M,768] @ [3072,768]^T -> xz fp32 [M,3072]
        gemm_bf16_wmma<<<dim3(MTOK/BM, (2*DINNER)/BN), 256, 0, stream>>>(
            xn_bf, inw_bf[dir], MTOK, 2*DINNER, DMODEL,
            xz_f, nullptr, nullptr, 0, nullptr, 0);

        // depthwise conv + SiLU -> xh (fp32 + bf16)
        conv_silu_kernel<<<EB, 256, 0, stream>>>(
            xz_f, conv_w[dir], conv_b[dir], xh_f, xh_bf, dir);

        // x_proj: [M,1536] @ [128(pad),1536]^T -> xdbl fp32 [M,128]
        gemm_bf16_wmma<<<dim3(MTOK/BM, XPN/BN), 256, 0, stream>>>(
            xh_bf, xprojw_bf[dir], MTOK, XPN, DINNER,
            xdbl_f, nullptr, nullptr, 0, nullptr, 0);

        // dt columns (0..47) -> padded bf16 [M,64]
        convert_pad_kernel<<<EB, 256, 0, stream>>>(
            xdbl_f, dtin_bf, MTOK, MTOK, XPN, DTRANK, DTPAD);

        // dt_proj + bias + softplus -> dt fp32 [M,1536]
        gemm_bf16_wmma<<<dim3(MTOK/BM, DINNER/BN), 256, 0, stream>>>(
            dtin_bf, dtw_bf[dir], MTOK, DINNER, DTPAD,
            dt_f, nullptr, dt_b[dir], 3, nullptr, 0);

        // chunked selective scan
        scan_phase1<<<SCAN_B, 256, 0, stream>>>(
            xh_f, dt_f, A_log[dir], xdbl_f, hend_f, ssum_f, dir);
        scan_phase2<<<(BATCH*DINNER + 255)/256, 256, 0, stream>>>(
            A_log[dir], hend_f, ssum_f, h0buf_f);
        scan_phase3<<<SCAN_B, 256, 0, stream>>>(
            xh_f, dt_f, A_log[dir], Dp[dir], xdbl_f, h0buf_f, y_f, dir);

        // y * silu(z) -> bf16
        gate_mul_kernel<<<EB, 256, 0, stream>>>(y_f, xz_f, ymul_bf);

        // out_proj: [M,1536] @ [768,1536]^T -> fo (accumulate for dir 1)
        gemm_bf16_wmma<<<dim3(MTOK/BM, DMODEL/BN), 256, 0, stream>>>(
            ymul_bf, outw_bf[dir], MTOK, DMODEL, DINNER,
            fo_f, nullptr, nullptr, 0, nullptr, dir);
    }

    // ---- 4. LN2 ----
    layernorm_kernel<<<MTOK, 256, 0, stream>>>(fo_f, ln2_g, ln2_b, m_bf, DMODEL);

    // ---- 5. FFN: gelu(m @ W1^T + b1) @ W2^T + b2 + x ----
    gemm_bf16_wmma<<<dim3(MTOK/BM, (4*DMODEL)/BN), 256, 0, stream>>>(
        m_bf, ffw1_bf, MTOK, 4*DMODEL, DMODEL,
        nullptr, h1_bf, ff_b1, 2, nullptr, 0);

    gemm_bf16_wmma<<<dim3(MTOK/BM, DMODEL/BN), 256, 0, stream>>>(
        h1_bf, ffw2_bf, MTOK, DMODEL, 4*DMODEL,
        outp, nullptr, ff_b2, 0, x, 0);
}
